// MessagePassingUnit_v1_56977036149411
// MI455X (gfx1250) — compile-verified, matrix operand order checked by reading, not run on hardware
//
#include <hip/hip_runtime.h>
#include <hip/hip_fp16.h>
#include <math.h>

// ---------------------------------------------------------------------------
// MI455X (gfx1250) fused kernel:
//   concat -> LayerNorm(1024) -> ReLU -> GEMM[1024x64] (v_wmma f16) ->
//   sigmoid -> mean(64) -> pair * gate * attn
// Memory-bound target ~33us (768MB @ 23.3TB/s); GEMM on WMMA to stay below
// the bandwidth roofline (VALU fp32 would be ~3x too slow for the 17.2 GFLOP).
// ---------------------------------------------------------------------------

typedef __attribute__((ext_vector_type(16))) _Float16 v16h;
typedef __attribute__((ext_vector_type(8)))  _Float16 h8;
typedef __attribute__((ext_vector_type(4)))  _Float16 h4;
typedef __attribute__((ext_vector_type(8)))  float    v8f;
typedef __attribute__((ext_vector_type(4)))  float    f4;

#define NROWS 131072
#define DDIM  512
#define TWOD  1024
#define FDIM  64
#define LNEPS 1e-5f

#define WAVES  4                    // waves per block (wave32)
#define RPW    16                   // rows per wave (one WMMA M-tile)
#define RPB    (WAVES * RPW)        // 64 rows per block
#define APAD   8                    // f16 pad per row: stride 2064B -> lane
                                    // row-strided b128 LDS reads hit distinct banks
#define AROW   (TWOD + APAD)        // 1032 halves per staged row

// ---------------------------------------------------------------------------
// Prep: lin_weight f32 [64][1024] -> f16 copy in workspace (L2-resident).
// ---------------------------------------------------------------------------
__global__ void wcvt_f16_kernel(const float* __restrict__ w,
                                _Float16* __restrict__ wh) {
    int i = (blockIdx.x * blockDim.x + threadIdx.x) * 4;
    f4 x = *(const f4*)(w + i);
    h4 h = { (_Float16)x.x, (_Float16)x.y, (_Float16)x.z, (_Float16)x.w };
    *(h4*)(wh + i) = h;
}

// ---------------------------------------------------------------------------
// Main fused kernel. Block = 128 threads (4 wave32), 64 rows per block.
// ---------------------------------------------------------------------------
__global__ __launch_bounds__(128)
void mpu_fused_kernel(const float* __restrict__ unary,
                      const float* __restrict__ pairt,
                      const float* __restrict__ attn,
                      const float* __restrict__ lnw_g,
                      const float* __restrict__ lnb_g,
                      const _Float16* __restrict__ Wh,   // [64][1024] f16
                      const float* __restrict__ linb,
                      float* __restrict__ out,           // [N][512]
                      float* __restrict__ gate_out) {    // [N]
    __shared__ _Float16 Ah[WAVES][RPW * AROW];   // staged concat rows, f16
    __shared__ float    lnw[TWOD];
    __shared__ float    lnb[TWOD];
    __shared__ float    sScale[WAVES][RPW];      // rsig
    __shared__ float    sShift[WAVES][RPW];      // -mu * rsig
    __shared__ float    sGate[WAVES][RPW];

    const int tid  = threadIdx.x;
    const int wave = tid >> 5;
    const int lane = tid & 31;
    const int l16  = lane & 15;
    const int hi   = lane >> 4;

    // --- cooperative: LayerNorm affine params into LDS -------------------
    for (int i = tid; i < TWOD; i += 128) { lnw[i] = lnw_g[i]; lnb[i] = lnb_g[i]; }

    // --- pass 1: stream 16 rows/wave, stats + f16 stage to LDS -----------
    const int rowBase = blockIdx.x * RPB + wave * RPW;
    for (int r = 0; r < RPW; ++r) {
        const int row = rowBase + r;
        const float* __restrict__ up = unary + (size_t)row * DDIM;
        const float* __restrict__ pp = pairt + (size_t)row * DDIM;
        // prefetch next row: 32 lanes x 64B blanket the full 2KB row
        // (lowered to global_prefetch_b8 on gfx1250)
        if (r + 1 < RPW) {
            __builtin_prefetch(up + DDIM + lane * 16, 0, 0);
            __builtin_prefetch(pp + DDIM + lane * 16, 0, 0);
        }
        _Float16* dst = &Ah[wave][r * AROW];
        float sum = 0.f, ssq = 0.f;
#pragma unroll
        for (int it = 0; it < 8; ++it) {
            const int c = it * 128 + lane * 4;   // concat column, coalesced 512B
            f4 x;
            if (it < 4) x = __builtin_nontemporal_load((const f4*)(up + c));
            else        x = *(const f4*)(pp + (c - DDIM));   // pair re-read later: keep RT
            sum += x.x + x.y + x.z + x.w;
            ssq += x.x * x.x + x.y * x.y + x.z * x.z + x.w * x.w;
            h4 h = { (_Float16)x.x, (_Float16)x.y, (_Float16)x.z, (_Float16)x.w };
            *(h4*)(dst + c) = h;                 // 8B LDS store, conflict-free
        }
#pragma unroll
        for (int m = 16; m >= 1; m >>= 1) {
            sum += __shfl_xor(sum, m, 32);
            ssq += __shfl_xor(ssq, m, 32);
        }
        if (lane == 0) {
            const float mu  = sum * (1.f / TWOD);
            const float var = ssq * (1.f / TWOD) - mu * mu;
            const float rs  = __builtin_amdgcn_rsqf(var + LNEPS);  // v_rsq_f32
            sScale[wave][r] = rs;
            sShift[wave][r] = -mu * rs;
        }
    }
    __syncthreads();

    // --- pass 2: fused LN-affine/ReLU fragment build + WMMA --------------
    // A layout (16-bit 16x32): lanes 0-15 row M=l16 hold K = kb..kb+7 and
    // kb+16..kb+23 with kb = kt*32 + hi*8 (hi selects the +8 K offset).
    const float myA = sScale[wave][l16];
    const float myB = sShift[wave][l16];
    const _Float16* aRow = &Ah[wave][l16 * AROW];

    v8f acc[4] = {};            // 4 N-tiles of 16 -> FD = 64
    for (int kt = 0; kt < 32; ++kt) {
        const int kb = kt * 32 + hi * 8;
        h8 xlo = *(const h8*)(aRow + kb);        // ds_load_b128
        h8 xhi = *(const h8*)(aRow + kb + 16);   // ds_load_b128
        v16h a;
#pragma unroll
        for (int j = 0; j < 8; ++j) {
            float v0 = fmaf((float)xlo[j], myA, myB);     // xhat
            v0 = fmaf(v0, lnw[kb + j], lnb[kb + j]);      // affine
            a[j] = (_Float16)fmaxf(v0, 0.f);              // relu
            float v1 = fmaf((float)xhi[j], myA, myB);
            v1 = fmaf(v1, lnw[kb + 16 + j], lnb[kb + 16 + j]);
            a[j + 8] = (_Float16)fmaxf(v1, 0.f);
        }
#pragma unroll
        for (int nt = 0; nt < 4; ++nt) {
            // B (32x16, K-major mirrored layout): lane holds column
            // n = nt*16 + l16, same K pattern as A. B = lin_weight[n][k].
            const _Float16* bp = Wh + (size_t)(nt * 16 + l16) * TWOD + kb;
            h8 blo = *(const h8*)bp;             // global b128, WGP$/L2-hot
            h8 bhi = *(const h8*)(bp + 16);
            v16h b = __builtin_shufflevector(blo, bhi,
                        0, 1, 2, 3, 4, 5, 6, 7, 8, 9, 10, 11, 12, 13, 14, 15);
            acc[nt] = __builtin_amdgcn_wmma_f32_16x16x32_f16(
                false, a, false, b, (short)0, acc[nt], false, false);
        }
    }

    // --- epilogue: bias + sigmoid + mean over 64 filters -----------------
    // C layout: VGPR r -> M=r (lanes 0-15) / M=r+8 (lanes 16-31), N=l16+nt*16.
    float biasn[4];
#pragma unroll
    for (int nt = 0; nt < 4; ++nt) biasn[nt] = linb[nt * 16 + l16];

#pragma unroll
    for (int r = 0; r < 8; ++r) {
        float s = 0.f;
#pragma unroll
        for (int nt = 0; nt < 4; ++nt) {
            float v = acc[nt][r] + biasn[nt];
            // sigmoid via v_exp_f32 + v_rcp_f32 (1-ulp approx is plenty;
            // avoids the exact-division v_div_scale/fixup sequence)
            s += __builtin_amdgcn_rcpf(1.f + __expf(-v));
        }
#pragma unroll
        for (int m = 1; m < 16; m <<= 1) s += __shfl_xor(s, m, 32);
        if (l16 == 0) {
            const int rr = r + hi * 8;
            const float g = s * (1.f / 64.f);
            sGate[wave][rr] = g;
            gate_out[rowBase + rr] = g;
        }
    }
    __syncthreads();

    // --- output: pair * gate * attn, coalesced NT stores -----------------
    for (int r = 0; r < RPW; ++r) {
        const int row = rowBase + r;
        const float ga = sGate[wave][r] * attn[row];
        const float* __restrict__ pp = pairt + (size_t)row * DDIM;
        float* __restrict__ op = out + (size_t)row * DDIM;
#pragma unroll
        for (int it = 0; it < 4; ++it) {
            const int c = it * 128 + lane * 4;
            f4 x = *(const f4*)(pp + c);          // L2-hot (streamed in pass 1)
            f4 y = { x.x * ga, x.y * ga, x.z * ga, x.w * ga };
            __builtin_nontemporal_store(y, (f4*)(op + c));
        }
    }
}

// ---------------------------------------------------------------------------
extern "C" void kernel_launch(void* const* d_in, const int* in_sizes, int n_in,
                              void* d_out, int out_size, void* d_ws, size_t ws_size,
                              hipStream_t stream) {
    (void)in_sizes; (void)n_in; (void)out_size; (void)ws_size;
    const float* unary = (const float*)d_in[0];
    const float* pairt = (const float*)d_in[1];
    const float* attn  = (const float*)d_in[2];
    const float* lnw   = (const float*)d_in[3];
    const float* lnb   = (const float*)d_in[4];
    const float* linw  = (const float*)d_in[5];
    const float* linb  = (const float*)d_in[6];

    float* out  = (float*)d_out;                  // [N*512] output
    float* gate = out + (size_t)NROWS * DDIM;     // [N] gate (tuple order)
    _Float16* wh = (_Float16*)d_ws;               // 128KB f16 weight copy

    // 64*1024 = 65536 elems, 4 per thread
    wcvt_f16_kernel<<<64, 256, 0, stream>>>(linw, wh);
    mpu_fused_kernel<<<NROWS / RPB, 128, 0, stream>>>(
        unary, pairt, attn, lnw, lnb, wh, linb, out, gate);
}
